// ViewNet_72584947302682
// MI455X (gfx1250) — matrix operand order repaired and checked by reading, compiled, not run on hardware
//
#include <hip/hip_runtime.h>

typedef __attribute__((ext_vector_type(16))) _Float16 v16h;
typedef __attribute__((ext_vector_type(8)))  float    v8f;

#define H_ 256
#define W_ 256
#define B_ 8
#define L_ 68
#define SD_ 0.01f

// ---- CDNA5 async global->LDS helpers (inline asm; CDNA5 ISA) ----
__device__ __forceinline__ unsigned lds_addr_of(const void* p) {
  // generic pointer to __shared__: low 32 bits are the wave-relative LDS offset
  return (unsigned)(unsigned long long)p;
}

__device__ __forceinline__ void async_ld_b32(unsigned lds, unsigned goff, const void* sbase) {
  // GVS mode: mem = SADDR(64) + VADDR(32) ; VDST = LDS byte address
  asm volatile("global_load_async_to_lds_b32 %0, %1, %2"
               :: "v"(lds), "v"(goff), "s"(sbase) : "memory");
}

__device__ __forceinline__ void wait_async0() {
  asm volatile("s_wait_asynccnt 0" ::: "memory");
}

// ---------------------------------------------------------------------------
// Conv3x3 "SAME" (zero pad), bias, optional ReLU, via WMMA f16 -> f32.
// One block (128 thr = 4 waves) handles one (batch b, output row y).
// LDS holds rows y-1..y+1 for all CIN channels, cols -1..256 (zero padded),
// plus a 32-float zero slot used to feed K-padding lanes branchlessly.
// Each wave computes 4 tiles of 16 pixels; per tile: KB wmma 16x16x32 steps.
// ---------------------------------------------------------------------------
template<int CIN, int COUT, bool RELU>
__global__ __launch_bounds__(128)
void conv3x3_wmma(const float* __restrict__ in, const float* __restrict__ w,
                  const float* __restrict__ bias, float* __restrict__ out)
{
  constexpr int KTOT = CIN * 9;
  constexpr int KB   = (KTOT + 31) / 32;
  constexpr int ZOFF = CIN * 3 * 260;          // zero-slot base (floats)
  __shared__ float sIn[CIN * 3 * 260 + 32];    // [ci][row 0..2][col 0..257(+pad)] + zeros

  const int y = blockIdx.x;
  const int b = blockIdx.y;

  // Zero slot for K-padding reads.
  if (threadIdx.x < 32) sIn[ZOFF + threadIdx.x] = 0.0f;

  // Cooperative LDS fill: interior via async global->LDS, borders zeroed.
  for (int t = threadIdx.x; t < CIN * 3 * 258; t += 128) {
    const int ci   = t / (3 * 258);
    const int rem  = t - ci * (3 * 258);
    const int r    = rem / 258;
    const int c    = rem - r * 258;
    const int phys = (ci * 3 + r) * 260 + c;
    const int yi   = y + r - 1;
    const int xi   = c - 1;
    if (yi >= 0 && yi < H_ && xi >= 0 && xi < W_) {
      const unsigned goff = (unsigned)((((b * CIN + ci) * H_ + yi) * W_ + xi) * 4);
      async_ld_b32(lds_addr_of(&sIn[phys]), goff, (const void*)in);
    } else {
      sIn[phys] = 0.0f;
    }
  }
  wait_async0();
  __syncthreads();

  const int lane = threadIdx.x & 31;
  const int wv   = threadIdx.x >> 5;
  const int n    = lane & 15;   // N (output channel) for B/C/D
  const int m    = lane & 15;   // M (pixel) for A
  const int hi   = lane >> 4;

  // B fragments: lane holds B[k = kb*32 + hi*16 + j][n] = w[n][ci][kh][kw].
  // Branchless: clamp index to 0 (always in-bounds), select 0 for pad lanes.
  v16h bf[KB];
#pragma unroll
  for (int kb = 0; kb < KB; ++kb) {
#pragma unroll
    for (int j = 0; j < 16; ++j) {
      const int  k     = kb * 32 + hi * 16 + j;
      const bool valid = (k < KTOT) && (n < COUT);
      const int  ci    = k / 9, tap = k - ci * 9;
      const int  idx   = valid ? ((n * CIN + ci) * 9 + tap) : 0;
      const float wvl  = w[idx];
      bf[kb][j] = valid ? (_Float16)wvl : (_Float16)0.0f;
    }
  }
  const float bv = (n < COUT) ? bias[n] : 0.0f;

  // 4 tiles of 16 pixels per wave (W = 16 tiles * 16 px)
#pragma unroll
  for (int ti = 0; ti < 4; ++ti) {
    const int x0 = (wv * 4 + ti) * 16;
    const int x  = x0 + m;
    v8f acc = {};
#pragma unroll
    for (int kb = 0; kb < KB; ++kb) {
      v16h a;
#pragma unroll
      for (int j = 0; j < 16; ++j) {
        const int kk = hi * 16 + j;
        const int k  = kb * 32 + kk;
        const int ci = k / 9, tap = k - ci * 9;
        const int kh = tap / 3, kw = tap - kh * 3;
        // Branchless: invalid (K-pad) elements read the zero slot.
        const int off = (k < KTOT) ? ((ci * 3 + kh) * 260 + x + kw)
                                   : (ZOFF + (j & 31));
        a[j] = (_Float16)sIn[off];
      }
      acc = __builtin_amdgcn_wmma_f32_16x16x32_f16(
          /*neg_a=*/false, a, /*neg_b=*/false, bf[kb],
          /*c_mod=*/(short)0, acc, /*reuse_a=*/false, /*reuse_b=*/false);
    }
    // Epilogue: lane's 8 results are pixels x0 + hi*8 + v, channel n
    float d[8];
#pragma unroll
    for (int v = 0; v < 8; ++v) {
      float t = acc[v] + bv;
      if (RELU) t = fmaxf(t, 0.0f);
      d[v] = t;
    }
    if (n < COUT) {
      float* orow = out + (((size_t)(b * COUT + n) * H_ + y) * W_ + x0 + hi * 8);
      *reinterpret_cast<float4*>(orow)     = make_float4(d[0], d[1], d[2], d[3]);
      *reinterpret_cast<float4*>(orow + 4) = make_float4(d[4], d[5], d[6], d[7]);
    }
  }
}

// ---------------------------------------------------------------------------
// Fused RBF flow field + grid build + bilinear border sample (8 channels).
// One thread per output pixel; landmarks + flow staged in LDS.
// ---------------------------------------------------------------------------
__global__ __launch_bounds__(256)
void warp_rbf(const float* __restrict__ refKey, const float* __restrict__ tarKey,
              const float* __restrict__ feats, float* __restrict__ out)
{
  __shared__ float tkx[L_], tky[L_], fxv[L_], fyv[L_];
  const int b   = blockIdx.y;
  const int tid = threadIdx.x;
  if (tid < L_) {
    const float rx = refKey[(b * L_ + tid) * 2 + 0];
    const float ry = refKey[(b * L_ + tid) * 2 + 1];
    const float tx = tarKey[(b * L_ + tid) * 2 + 0];
    const float ty = tarKey[(b * L_ + tid) * 2 + 1];
    tkx[tid] = tx; tky[tid] = ty; fxv[tid] = tx - rx; fyv[tid] = ty - ry;
  }
  __syncthreads();

  const int pix = blockIdx.x * 256 + tid;
  const int y = pix >> 8;      // W_ == 256
  const int x = pix & 255;
  const float mx = (float)x * (1.0f / (W_ - 1));
  const float my = (float)y * (1.0f / (H_ - 1));
  const float inv2sd2 = 1.0f / (2.0f * SD_ * SD_);

  float mlw = -3.4e38f;
#pragma unroll 4
  for (int i = 0; i < L_; ++i) {
    const float dx = mx - tkx[i], dy = my - tky[i];
    mlw = fmaxf(mlw, -(dx * dx + dy * dy) * inv2sd2);
  }
  float S = 0.0f, FX = 0.0f, FY = 0.0f;
#pragma unroll 4
  for (int i = 0; i < L_; ++i) {
    const float dx = mx - tkx[i], dy = my - tky[i];
    const float e = __expf(-(dx * dx + dy * dy) * inv2sd2 - mlw);
    S += e; FX += e * fxv[i]; FY += e * fyv[i];
  }
  const float Xf = 2.0f * (FX / S);     // S >= 1 (max term), no NaN
  const float Yf = 2.0f * (FY / S);
  const float gx = (2.0f * mx - 1.0f) - Xf;
  const float gy = (2.0f * my - 1.0f) - Yf;

  float ix = fminf(fmaxf((gx + 1.0f) * 0.5f * (W_ - 1), 0.0f), (float)(W_ - 1));
  float iy = fminf(fmaxf((gy + 1.0f) * 0.5f * (H_ - 1), 0.0f), (float)(H_ - 1));
  const float x0f = floorf(ix), y0f = floorf(iy);
  const float wx = ix - x0f, wy = iy - y0f;
  const int x0 = (int)x0f, y0 = (int)y0f;
  const int x1 = min(x0 + 1, W_ - 1), y1 = min(y0 + 1, H_ - 1);

  const float w00 = (1.0f - wx) * (1.0f - wy), w01 = wx * (1.0f - wy);
  const float w10 = (1.0f - wx) * wy,          w11 = wx * wy;
#pragma unroll
  for (int c = 0; c < 8; ++c) {
    const float* fp = feats + (size_t)(b * 8 + c) * H_ * W_;
    const float v = fp[y0 * W_ + x0] * w00 + fp[y0 * W_ + x1] * w01 +
                    fp[y1 * W_ + x0] * w10 + fp[y1 * W_ + x1] * w11;
    out[((size_t)(b * 8 + c) * H_ + y) * W_ + x] = v;
  }
}

// ---------------------------------------------------------------------------
extern "C" void kernel_launch(void* const* d_in, const int* in_sizes, int n_in,
                              void* d_out, int out_size, void* d_ws, size_t ws_size,
                              hipStream_t stream)
{
  (void)in_sizes; (void)n_in; (void)out_size; (void)ws_size;
  const float* refKey = (const float*)d_in[2];
  const float* tarKey = (const float*)d_in[3];
  const float* x      = (const float*)d_in[4];
  const float *wa[6], *ba[6], *wb[6], *bb[6];
  for (int i = 0; i < 6; ++i) {
    wa[i] = (const float*)d_in[5  + 2 * i];
    ba[i] = (const float*)d_in[6  + 2 * i];
    wb[i] = (const float*)d_in[17 + 2 * i];
    bb[i] = (const float*)d_in[18 + 2 * i];
  }

  float* p0 = (float*)d_ws;
  float* p1 = p0 + (size_t)B_ * 8 * H_ * W_;

  const dim3 cg(H_, B_), cb(128);
  conv3x3_wmma<3, 8, true ><<<cg, cb, 0, stream>>>(x,  wa[0], ba[0], p0);
  conv3x3_wmma<8, 8, true ><<<cg, cb, 0, stream>>>(p0, wa[1], ba[1], p1);
  conv3x3_wmma<8, 8, true ><<<cg, cb, 0, stream>>>(p1, wa[2], ba[2], p0);
  conv3x3_wmma<8, 8, true ><<<cg, cb, 0, stream>>>(p0, wa[3], ba[3], p1);
  conv3x3_wmma<8, 8, true ><<<cg, cb, 0, stream>>>(p1, wa[4], ba[4], p0);
  conv3x3_wmma<8, 8, false><<<cg, cb, 0, stream>>>(p0, wa[5], ba[5], p1);  // feats

  const dim3 wg((H_ * W_) / 256, B_), wbk(256);
  warp_rbf<<<wg, wbk, 0, stream>>>(refKey, tarKey, p1, p0);                // warped

  conv3x3_wmma<8, 8, true ><<<cg, cb, 0, stream>>>(p0, wb[0], bb[0], p1);
  conv3x3_wmma<8, 8, true ><<<cg, cb, 0, stream>>>(p1, wb[1], bb[1], p0);
  conv3x3_wmma<8, 8, true ><<<cg, cb, 0, stream>>>(p0, wb[2], bb[2], p1);
  conv3x3_wmma<8, 8, true ><<<cg, cb, 0, stream>>>(p1, wb[3], bb[3], p0);
  conv3x3_wmma<8, 8, true ><<<cg, cb, 0, stream>>>(p0, wb[4], bb[4], p1);
  conv3x3_wmma<8, 3, false><<<cg, cb, 0, stream>>>(p1, wb[5], bb[5], (float*)d_out);
}
